// AttentionBlock_65481071402025
// MI455X (gfx1250) — matrix operand order repaired and checked by reading, not compiled, not run on hardware
//
#include <hip/hip_runtime.h>

typedef _Float16 half16 __attribute__((ext_vector_type(16)));
typedef _Float16 half8  __attribute__((ext_vector_type(8)));
typedef _Float16 half2v __attribute__((ext_vector_type(2)));
typedef float    f32x8  __attribute__((ext_vector_type(8)));

#define NTOK 8192
#define DIM  64

// ---------------- Pass 1: Q = (x@R)/8, K = x@E (f16), xT = x^T (f16) -------
__global__ __launch_bounds__(64) void qk_pack_kernel(
    const float* __restrict__ x, const float* __restrict__ R,
    const float* __restrict__ E, _Float16* __restrict__ Qh,
    _Float16* __restrict__ Kh, _Float16* __restrict__ xT)
{
  __shared__ float xrow[DIM];
  const int r = blockIdx.x;
  const int t = threadIdx.x;
  const float xv = x[r * DIM + t];
  xrow[t] = xv;
  __syncthreads();
  float q = 0.f, k = 0.f;
#pragma unroll
  for (int kk = 0; kk < DIM; ++kk) {
    const float xk = xrow[kk];
    q = fmaf(xk, R[kk * DIM + t], q);
    k = fmaf(xk, E[kk * DIM + t], k);
  }
  Qh[r * DIM + t] = (_Float16)(q * 0.125f);   // fold 1/sqrt(64) into Q
  Kh[r * DIM + t] = (_Float16)k;
  xT[t * NTOK + r] = (_Float16)xv;            // V^T for the P@V WMMA
}

// ---------------- helpers ---------------------------------------------------
static __device__ inline half16 cat8(half8 a, half8 b) {
  return __builtin_shufflevector(a, b, 0,1,2,3,4,5,6,7,8,9,10,11,12,13,14,15);
}
static __device__ inline unsigned pack2(float a, float b) {
#if __has_builtin(__builtin_amdgcn_cvt_pkrtz)
  return __builtin_bit_cast(unsigned, __builtin_amdgcn_cvt_pkrtz(a, b));
#else
  half2v pk; pk.x = (_Float16)a; pk.y = (_Float16)b;
  return __builtin_bit_cast(unsigned, pk);
#endif
}
// lane <-> lane^16 exchange within wave32: single VALU v_permlanex16_b32
// (identity nibble selects read lane i of the *other* 16-lane half),
// instead of ds_bpermute + s_wait_dscnt round-trips through LDS.
static __device__ inline int xor16_i(int v) {
#if __has_builtin(__builtin_amdgcn_permlanex16)
  return __builtin_amdgcn_permlanex16(v, v, 0x76543210, 0xfedcba98, false, false);
#else
  return __shfl_xor(v, 16, 32);
#endif
}
static __device__ inline float xor16_f(float v) {
  return __builtin_bit_cast(float, xor16_i(__builtin_bit_cast(int, v)));
}
static __device__ inline f32x8 wmma_f16(half16 a, half16 b, f32x8 c) {
  return __builtin_amdgcn_wmma_f32_16x16x32_f16(false, a, false, b,
                                                (short)0, c, false, false);
}

// ---------------- Pass 2: flash attention, S^T / O^T formulation -----------
// Each wave owns 16 query rows. Per 32-key iteration:
//   S^T = K_tile(16x64) @ Q^T(64x16)   -> 2 C-tiles, lane = query column
//   online softmax: per-lane reduce + one permlanex16
//   O^T(64x16) += V^T(64x32) @ P^T(32x16) -> 4 C-tiles (dim blocks)
__global__ __launch_bounds__(128) void flash_kernel(
    const _Float16* __restrict__ Qh, const _Float16* __restrict__ Kh,
    const _Float16* __restrict__ xT, float* __restrict__ out)
{
  const int lane = threadIdx.x & 31;
  const int wave = threadIdx.x >> 5;
  const int rowbase = (blockIdx.x * 4 + wave) * 16;
  const int j  = lane & 15;   // query column within tile
  const int lh = lane >> 4;   // lane half

  // Q^T as WMMA B operand (K = 0..31 and 32..63), loaded once, reused 256x.
  const half16 bq0 = cat8(*(const half8*)(Qh + (rowbase + j) * DIM + lh * 8),
                          *(const half8*)(Qh + (rowbase + j) * DIM + 16 + lh * 8));
  const half16 bq1 = cat8(*(const half8*)(Qh + (rowbase + j) * DIM + 32 + lh * 8),
                          *(const half8*)(Qh + (rowbase + j) * DIM + 48 + lh * 8));

  f32x8 ot[4] = {{}, {}, {}, {}};   // O^T accumulators: 4 dim-tiles of 16
  float m = -3.402823e38f;          // running row max (per query column)
  float l = 0.f;                    // running row sum

  for (int n = 0; n < NTOK; n += 32) {
    // ---- A operands: K rows (two key sub-tiles, two K-chunks each) ----
    const _Float16* krow0 = Kh + (n + j) * DIM;
    const _Float16* krow1 = Kh + (n + 16 + j) * DIM;
    const half16 ak00 = cat8(*(const half8*)(krow0 + lh * 8),
                             *(const half8*)(krow0 + 16 + lh * 8));
    const half16 ak01 = cat8(*(const half8*)(krow0 + 32 + lh * 8),
                             *(const half8*)(krow0 + 48 + lh * 8));
    const half16 ak10 = cat8(*(const half8*)(krow1 + lh * 8),
                             *(const half8*)(krow1 + 16 + lh * 8));
    const half16 ak11 = cat8(*(const half8*)(krow1 + 32 + lh * 8),
                             *(const half8*)(krow1 + 48 + lh * 8));

    const f32x8 zero = {};
    f32x8 st0 = wmma_f16(ak00, bq0, zero);   // keys n..n+15   x queries
    st0       = wmma_f16(ak01, bq1, st0);
    f32x8 st1 = wmma_f16(ak10, bq0, zero);   // keys n+16..n+31
    st1       = wmma_f16(ak11, bq1, st1);

    // ---- online softmax (per lane: 16 keys; other 16 in partner lane) ----
    float tmax = st0[0];
#pragma unroll
    for (int v = 1; v < 8; ++v) tmax = fmaxf(tmax, st0[v]);
#pragma unroll
    for (int v = 0; v < 8; ++v) tmax = fmaxf(tmax, st1[v]);
    tmax = fmaxf(tmax, xor16_f(tmax));
    const float mnew  = fmaxf(m, tmax);
    const float alpha = __expf(m - mnew);

    f32x8 p0, p1;
    float rsum = 0.f;
#pragma unroll
    for (int v = 0; v < 8; ++v) { p0[v] = __expf(st0[v] - mnew); rsum += p0[v]; }
#pragma unroll
    for (int v = 0; v < 8; ++v) { p1[v] = __expf(st1[v] - mnew); rsum += p1[v]; }
    rsum += xor16_f(rsum);
    m = mnew;
    l = l * alpha + rsum;

#pragma unroll
    for (int v = 0; v < 8; ++v) {
      ot[0][v] *= alpha; ot[1][v] *= alpha; ot[2][v] *= alpha; ot[3][v] *= alpha;
    }

    // ---- P^T -> WMMA B operand (f16 packs + one cross-half exchange) ----
    unsigned own0[4], own1[4], oth0[4], oth1[4];
#pragma unroll
    for (int v = 0; v < 4; ++v) {
      own0[v] = pack2(p0[2 * v], p0[2 * v + 1]);
      own1[v] = pack2(p1[2 * v], p1[2 * v + 1]);
    }
#pragma unroll
    for (int v = 0; v < 4; ++v) {
      oth0[v] = (unsigned)xor16_i((int)own0[v]);
      oth1[v] = (unsigned)xor16_i((int)own1[v]);
    }
    union { unsigned u[8]; half16 h; } bp;
#pragma unroll
    for (int v = 0; v < 4; ++v) {
      bp.u[v]     = (lh == 0) ? own0[v] : oth1[v];  // keys 0..7  / 16..23
      bp.u[v + 4] = (lh == 0) ? oth0[v] : own1[v];  // keys 8..15 / 24..31
    }

    // ---- O^T += V^T @ P^T  (V^T rows are contiguous in xT) ----
#pragma unroll
    for (int t = 0; t < 4; ++t) {
      const _Float16* vrow = xT + (16 * t + j) * NTOK + n;
      const half16 av = cat8(*(const half8*)(vrow + lh * 16),
                             *(const half8*)(vrow + lh * 16 + 8));
      ot[t] = wmma_f16(av, bp.h, ot[t]);
    }
  }

  // ---- normalize and write: lane j owns query column rowbase+j ----
  const float invl = 1.0f / l;
#pragma unroll
  for (int t = 0; t < 4; ++t)
#pragma unroll
    for (int v = 0; v < 8; ++v)
      out[(rowbase + j) * DIM + 16 * t + lh * 8 + v] = ot[t][v] * invl;
}

extern "C" void kernel_launch(void* const* d_in, const int* in_sizes, int n_in,
                              void* d_out, int out_size, void* d_ws, size_t ws_size,
                              hipStream_t stream) {
  const float* x = (const float*)d_in[0];
  const float* R = (const float*)d_in[1];
  const float* E = (const float*)d_in[2];
  float* out = (float*)d_out;

  _Float16* Qh = (_Float16*)d_ws;          // 1 MB
  _Float16* Kh = Qh + NTOK * DIM;          // 1 MB
  _Float16* xT = Kh + NTOK * DIM;          // 1 MB
  (void)in_sizes; (void)n_in; (void)out_size; (void)ws_size;

  qk_pack_kernel<<<NTOK, DIM, 0, stream>>>(x, R, E, Qh, Kh, xT);
  flash_kernel<<<NTOK / 64, 128, 0, stream>>>(Qh, Kh, xT, out);
}
